// CrossAttentionLayer_16707422781729
// MI455X (gfx1250) — matrix operand order repaired and checked by reading, not compiled, not run on hardware
//
#include <hip/hip_runtime.h>
#include <hip/hip_bf16.h>

#define N_SEQ   2048
#define D_MODEL 1024
#define NHEADS  16
#define DK      64

typedef __attribute__((ext_vector_type(16))) __bf16 v16bf;
typedef __attribute__((ext_vector_type(8)))  __bf16 v8bf;
typedef __attribute__((ext_vector_type(8)))  float  v8f;

static __device__ __forceinline__ v8f wmma_bf16(v16bf a, v16bf b, v8f c) {
    // (neg_a, A, neg_b, B, c_mod, C, reuse_a, reuse_b)
    return __builtin_amdgcn_wmma_f32_16x16x32_bf16(false, a, false, b, (short)0, c,
                                                   false, false);
}

static __device__ __forceinline__ v8bf ld8(const __bf16* p) {
    return *(const v8bf*)p;       // 16B load, all call sites 16B-aligned
}

static __device__ __forceinline__ v16bf cat(v8bf lo, v8bf hi) {
    return __builtin_shufflevector(lo, hi, 0, 1, 2, 3, 4, 5, 6, 7,
                                   8, 9, 10, 11, 12, 13, 14, 15);
}

// ---------------------------------------------------------------------------
// One-shot f32 -> bf16 conversion of tgt[0], memory[0], Wq, Wk, Wv.
// grid.y selects the array; each thread converts 8 contiguous elements.
// ---------------------------------------------------------------------------
__global__ void cvt_kernel(const float* __restrict__ tgt,
                           const float* __restrict__ mem,
                           const float* __restrict__ wq,
                           const float* __restrict__ wk,
                           const float* __restrict__ wv,
                           __bf16* __restrict__ dxq,
                           __bf16* __restrict__ dxm,
                           __bf16* __restrict__ dwq,
                           __bf16* __restrict__ dwk,
                           __bf16* __restrict__ dwv) {
    const int z = blockIdx.y;
    const float* src;
    __bf16* dst;
    size_t n;
    switch (z) {
        case 0:  src = tgt; dst = dxq; n = (size_t)N_SEQ * D_MODEL;   break;
        case 1:  src = mem; dst = dxm; n = (size_t)N_SEQ * D_MODEL;   break;
        case 2:  src = wq;  dst = dwq; n = (size_t)D_MODEL * D_MODEL; break;
        case 3:  src = wk;  dst = dwk; n = (size_t)D_MODEL * D_MODEL; break;
        default: src = wv;  dst = dwv; n = (size_t)D_MODEL * D_MODEL; break;
    }
    const size_t i = ((size_t)blockIdx.x * blockDim.x + threadIdx.x) * 8;
    if (i >= n) return;
    const float4 f0 = ((const float4*)(src + i))[0];
    const float4 f1 = ((const float4*)(src + i))[1];
    v8bf r;
    r[0] = (__bf16)f0.x; r[1] = (__bf16)f0.y; r[2] = (__bf16)f0.z; r[3] = (__bf16)f0.w;
    r[4] = (__bf16)f1.x; r[5] = (__bf16)f1.y; r[6] = (__bf16)f1.z; r[7] = (__bf16)f1.w;
    *(v8bf*)(dst + i) = r;
}

// ---------------------------------------------------------------------------
// Projections: C[m,n] = sum_k X[m,k] * W[n,k]   (y = x @ W^T, NT-GEMM, bf16 in)
// One wave -> 16x64 tile. Ping-pong double-buffered k-loop (step 64 = 2 WMMA
// k-slices): slot-B loads issue before slot-A WMMAs; all B-operand addresses
// are one base pointer + immediate offsets (c*32KB fits the 24-bit ioffset).
// which: 0 -> Q (pre-scaled by 1/sqrt(dk)), 1 -> K, 2 -> V stored transposed.
// ---------------------------------------------------------------------------
__global__ void proj_kernel(const __bf16* __restrict__ Xq,
                            const __bf16* __restrict__ Xm,
                            const __bf16* __restrict__ Wqb,
                            const __bf16* __restrict__ Wkb,
                            const __bf16* __restrict__ Wvb,
                            __bf16* __restrict__ Qs,
                            __bf16* __restrict__ Ks,
                            __bf16* __restrict__ Vts) {
    const int m0    = blockIdx.x * 16;
    const int n0    = blockIdx.y * 64;
    const int which = blockIdx.z;
    const __bf16* X = (which == 0) ? Xq : Xm;
    const __bf16* W = (which == 0) ? Wqb : ((which == 1) ? Wkb : Wvb);

    const int l  = threadIdx.x;
    const int lm = l & 15;
    const int lh = l >> 4;

    // A-operand: row m0+lm, per-lane k elems {k0+lh*8+0..7, k0+16+lh*8+0..7}
    const __bf16* xptr = X + (size_t)(m0 + lm) * D_MODEL + lh * 8;
    // B-operand: column n0+c*16+lm, k elems {k0+lh*8+0..7, k0+16+lh*8+0..7}
    // column-group offset c*16*D_MODEL folds into the load immediate.
    const __bf16* wptr = W + (size_t)(n0 + lm) * D_MODEL + lh * 8;

    v8f acc[4] = {};

    v8bf aL0, aH0, bL0[4], bH0[4];   // slot 0
    v8bf aL1, aH1, bL1[4], bH1[4];   // slot 1

    // prologue: slot 0 <- k0 = 0
    aL0 = ld8(xptr);
    aH0 = ld8(xptr + 16);
#pragma unroll
    for (int c = 0; c < 4; c++) {
        bL0[c] = ld8(wptr + c * 16 * D_MODEL);
        bH0[c] = ld8(wptr + c * 16 * D_MODEL + 16);
    }

    for (int k0 = 0; k0 < D_MODEL; k0 += 64) {
        // slot 1 <- k0 + 32 (always in range: D_MODEL % 64 == 0)
        aL1 = ld8(xptr + k0 + 32);
        aH1 = ld8(xptr + k0 + 48);
#pragma unroll
        for (int c = 0; c < 4; c++) {
            bL1[c] = ld8(wptr + k0 + 32 + c * 16 * D_MODEL);
            bH1[c] = ld8(wptr + k0 + 32 + c * 16 * D_MODEL + 16);
        }
        // consume slot 0
        {
            const v16bf a = cat(aL0, aH0);
#pragma unroll
            for (int c = 0; c < 4; c++)
                acc[c] = wmma_bf16(a, cat(bL0[c], bH0[c]), acc[c]);
        }
        // slot 0 <- k0 + 64
        if (k0 + 64 < D_MODEL) {
            aL0 = ld8(xptr + k0 + 64);
            aH0 = ld8(xptr + k0 + 80);
#pragma unroll
            for (int c = 0; c < 4; c++) {
                bL0[c] = ld8(wptr + k0 + 64 + c * 16 * D_MODEL);
                bH0[c] = ld8(wptr + k0 + 64 + c * 16 * D_MODEL + 16);
            }
        }
        // consume slot 1
        {
            const v16bf a = cat(aL1, aH1);
#pragma unroll
            for (int c = 0; c < 4; c++)
                acc[c] = wmma_bf16(a, cat(bL1[c], bH1[c]), acc[c]);
        }
    }

    // C/D layout: vgpr j -> row m0 + j + 8*lh, col n0 + c*16 + lm
#pragma unroll
    for (int c = 0; c < 4; c++) {
#pragma unroll
        for (int j = 0; j < 8; j++) {
            const int m = m0 + j + 8 * lh;
            const int n = n0 + c * 16 + lm;
            const float v = acc[c][j];
            if (which == 0)
                Qs[(size_t)m * D_MODEL + n] = (__bf16)(v * 0.125f);  // 1/sqrt(dk)
            else if (which == 1)
                Ks[(size_t)m * D_MODEL + n] = (__bf16)v;
            else
                Vts[(size_t)n * N_SEQ + m] = (__bf16)v;              // V transposed
        }
    }
}

// ---------------------------------------------------------------------------
// Flash-style attention, one wave per (16-query tile, head).
// scores^T = K_h x Q_h^T: each lane owns one query column -> per-lane softmax
// stats + one shfl_xor(16). Then att^T += Vt-chunk(16x32) x P^T(32x16).
// Two incremented base pointers; all fragment offsets are load immediates.
// ---------------------------------------------------------------------------
__global__ void attn_kernel(const __bf16* __restrict__ Qs,
                            const __bf16* __restrict__ Ks,
                            const __bf16* __restrict__ Vts,
                            float* __restrict__ att0) {
    const int q0 = blockIdx.x * 16;
    const int h  = blockIdx.y;
    const int l  = threadIdx.x;
    const int lm = l & 15;
    const int lh = l >> 4;

    // Q as B-operand (column = query q0+lm), k-slices [0,32) and [32,64)
    const __bf16* qrow = Qs + (size_t)(q0 + lm) * D_MODEL + h * DK;
    const v16bf bq0 = cat(ld8(qrow + lh * 8), ld8(qrow + lh * 8 + 16));
    const v16bf bq1 = cat(ld8(qrow + 32 + lh * 8), ld8(qrow + 32 + lh * 8 + 16));

    // K rows: key n0+lm at kptr (fragment1 = +16*D_MODEL, a 32KB immediate)
    const __bf16* kptr = Ks + (size_t)lm * D_MODEL + h * DK + lh * 8;
    // Vt rows: d = h*DK + c*16 + lm; chunk offset c*16*N_SEQ = c*64KB immediate
    const __bf16* vptr = Vts + (size_t)(h * DK + lm) * N_SEQ + lh * 16;

    float rowm = -__builtin_inff();   // running max  (per query lm, replicated)
    float rowl = 0.0f;                // running sum
    v8f   o[4] = {};                  // att^T accumulators, 4 chunks of 16 d's

    for (int n0 = 0; n0 < N_SEQ; n0 += 32) {
        // ---- issue ALL loads for this tile first (one wait, full MLP) ----
        const v8bf k00 = ld8(kptr),      k01 = ld8(kptr + 16);
        const v8bf k02 = ld8(kptr + 32), k03 = ld8(kptr + 48);
        const v8bf k10 = ld8(kptr + 16 * D_MODEL),      k11 = ld8(kptr + 16 * D_MODEL + 16);
        const v8bf k12 = ld8(kptr + 16 * D_MODEL + 32), k13 = ld8(kptr + 16 * D_MODEL + 48);
        v8bf vlo[4], vhi[4];
#pragma unroll
        for (int c = 0; c < 4; c++) {
            vlo[c] = ld8(vptr + c * 16 * N_SEQ);
            vhi[c] = ld8(vptr + c * 16 * N_SEQ + 8);
        }
        if (n0 + 32 < N_SEQ) {   // uniform branch; prefetch next K tile
            __builtin_prefetch(kptr + (size_t)32 * D_MODEL, 0, 0);
            __builtin_prefetch(kptr + (size_t)48 * D_MODEL, 0, 0);
        }

        // ---- scores^T: two 16(key)x16(query) fragments, K=64 -> 2 WMMAs each
        v8f s0 = {}, s1 = {};
        s0 = wmma_bf16(cat(k00, k01), bq0, s0);
        s0 = wmma_bf16(cat(k02, k03), bq1, s0);
        s1 = wmma_bf16(cat(k10, k11), bq0, s1);
        s1 = wmma_bf16(cat(k12, k13), bq1, s1);

        // ---- online softmax (query = lane%16; keys split lane vs lane^16) --
        float tm = s0[0];
#pragma unroll
        for (int j = 0; j < 8; j++) tm = fmaxf(tm, fmaxf(s0[j], s1[j]));
        tm = fmaxf(tm, __shfl_xor(tm, 16, 32));
        const float mnew = fmaxf(rowm, tm);
        const float corr = __expf(rowm - mnew);   // first iter: exp(-inf)=0

        float p0[8], p1[8];
        float psum = 0.0f;
#pragma unroll
        for (int j = 0; j < 8; j++) {
            p0[j] = __expf(s0[j] - mnew);
            p1[j] = __expf(s1[j] - mnew);
            psum += p0[j] + p1[j];
        }
        psum += __shfl_xor(psum, 16, 32);
        rowl = rowl * corr + psum;
        rowm = mnew;

        // ---- build P^T B-operand: lane<16 needs keys n0+0..15, lane>=16 keys
        // n0+16..31; half comes from partner lane (l^16) via shfl_xor ----
        v16bf pb;
#pragma unroll
        for (int j = 0; j < 8; j++) {
            const float send = lh ? p0[j] : p1[j];
            const float recv = __shfl_xor(send, 16, 32);
            const float own  = lh ? p1[j] : p0[j];
            pb[j]     = (__bf16)(lh ? recv : own);
            pb[j + 8] = (__bf16)(lh ? own : recv);
        }

        // ---- att^T accumulate: o = o*corr + Vt(16x32) x P^T(32x16) ----
#pragma unroll
        for (int c = 0; c < 4; c++) {
#pragma unroll
            for (int j = 0; j < 8; j++) o[c][j] *= corr;   // corr uniform per lane
            o[c] = wmma_bf16(cat(vlo[c], vhi[c]), pb, o[c]);
        }

        kptr += (size_t)32 * D_MODEL;
        vptr += 32;
    }

    const float inv = 1.0f / rowl;
#pragma unroll
    for (int c = 0; c < 4; c++) {
#pragma unroll
        for (int j = 0; j < 8; j++) {
            att0[(size_t)(q0 + lm) * D_MODEL + h * DK + c * 16 + j + 8 * lh] =
                o[c][j] * inv;
        }
    }
}

// ---------------------------------------------------------------------------
// out[b,n,d] = tgt[b,n,d] + att0[n,d]   (broadcast over batch)
// ---------------------------------------------------------------------------
__global__ void add_bcast_kernel(const float* __restrict__ tgt,
                                 const float* __restrict__ att0,
                                 float* __restrict__ out) {
    const size_t i     = (size_t)blockIdx.x * blockDim.x + threadIdx.x;  // float4 idx
    const size_t plane = (size_t)N_SEQ * D_MODEL / 4;
    const float4 a = ((const float4*)att0)[i];
#pragma unroll
    for (int b = 0; b < 4; b++) {
        float4 t = ((const float4*)tgt)[b * plane + i];
        float4 r;
        r.x = t.x + a.x; r.y = t.y + a.y; r.z = t.z + a.z; r.w = t.w + a.w;
        ((float4*)out)[b * plane + i] = r;
    }
}

extern "C" void kernel_launch(void* const* d_in, const int* in_sizes, int n_in,
                              void* d_out, int out_size, void* d_ws, size_t ws_size,
                              hipStream_t stream) {
    const float* tgt = (const float*)d_in[0];
    const float* mem = (const float*)d_in[1];
    const float* Wq  = (const float*)d_in[2];
    const float* Wk  = (const float*)d_in[3];
    const float* Wv  = (const float*)d_in[4];
    float* out = (float*)d_out;

    // workspace (bf16 unless noted):
    // Xq 4MB | Xm 4MB | Wqb 2MB | Wkb 2MB | Wvb 2MB | Q 4MB | K 4MB | Vt 4MB |
    // att0 (f32) 8MB    => ~34MB total
    const size_t XN = (size_t)N_SEQ * D_MODEL;
    const size_t WN = (size_t)D_MODEL * D_MODEL;
    __bf16* Xqb  = (__bf16*)d_ws;
    __bf16* Xmb  = Xqb + XN;
    __bf16* Wqb  = Xmb + XN;
    __bf16* Wkb  = Wqb + WN;
    __bf16* Wvb  = Wkb + WN;
    __bf16* Qs   = Wvb + WN;
    __bf16* Ks   = Qs + XN;
    __bf16* Vts  = Ks + XN;
    float*  att0 = (float*)(Vts + XN);

    // 1) convert inputs to bf16 (batch 0 planes of tgt/memory, plus weights)
    {
        const int blocks_x = (int)((XN / 8 + 255) / 256);
        cvt_kernel<<<dim3(blocks_x, 5), 256, 0, stream>>>(
            tgt, mem, Wq, Wk, Wv, Xqb, Xmb, Wqb, Wkb, Wvb);
    }
    // 2) Q/K/V projections (bf16 WMMA, f32 accumulate)
    proj_kernel<<<dim3(N_SEQ / 16, D_MODEL / 64, 3), 32, 0, stream>>>(
        Xqb, Xmb, Wqb, Wkb, Wvb, Qs, Ks, Vts);
    // 3) flash attention over batch 0
    attn_kernel<<<dim3(N_SEQ / 16, NHEADS), 32, 0, stream>>>(Qs, Ks, Vts, att0);
    // 4) broadcast residual add
    add_bcast_kernel<<<(N_SEQ * D_MODEL / 4) / 256, 256, 0, stream>>>(tgt, att0, out);
}
